// FlashAttention_23270132809797
// MI455X (gfx1250) — matrix operand order repaired and checked by reading, not compiled, not run on hardware
//
#include <hip/hip_runtime.h>
#include <hip/hip_bf16.h>

typedef __bf16 bf16_t;
typedef __attribute__((ext_vector_type(16))) __bf16 v16bf;
typedef __attribute__((ext_vector_type(8)))  __bf16 v8bf;
typedef __attribute__((ext_vector_type(8)))  float  v8f;
typedef int v4i __attribute__((vector_size(16)));
typedef __attribute__((address_space(1))) v4i* gv4i_ptr;
typedef __attribute__((address_space(3))) v4i* lv4i_ptr;

#define N_CTX   8192
#define DHEAD   64
#define WAVES   4
#define NTHREADS (32 * WAVES)        // 128
#define BLOCK_Q  (16 * WAVES)        // 64 Q rows per block
#define BC      64                   // KV tile rows
#define KP      72                   // padded LDS row (bf16 elems), 144B = 9*16B
#define NT      (N_CTX / BC)         // 128 KV tiles

// ---------------- async global->LDS copy (gfx1250 path with fallback) --------
__device__ __forceinline__ void cp_async16(const bf16_t* gsrc, bf16_t* ldst) {
#if __has_builtin(__builtin_amdgcn_global_load_async_to_lds_b128)
    __builtin_amdgcn_global_load_async_to_lds_b128(
        (gv4i_ptr)gsrc, (lv4i_ptr)ldst, 0, 0);
#else
    *(v8bf*)ldst = *(const v8bf*)gsrc;
#endif
}
__device__ __forceinline__ void wait_async_all() {
#if __has_builtin(__builtin_amdgcn_global_load_async_to_lds_b128)
#if __has_builtin(__builtin_amdgcn_s_wait_asynccnt)
    __builtin_amdgcn_s_wait_asynccnt(0);
#else
    asm volatile("s_wait_asynccnt 0" ::: "memory");
#endif
#endif
}

// ---------------- raw v_exp_f32 (flush-to-zero semantics are fine here) ------
__device__ __forceinline__ float fast_exp2(float x) {
#if __has_builtin(__builtin_amdgcn_exp2f)
    return __builtin_amdgcn_exp2f(x);
#else
    return exp2f(x);
#endif
}

// ---------------- xor-butterfly max over the 16-lane half-group --------------
__device__ __forceinline__ float rowmax_bfly16(float x) {
#if __has_builtin(__builtin_amdgcn_permlane16)
    unsigned u, p;
    u = __float_as_uint(x);  // xor 1
    p = __builtin_amdgcn_permlane16(u, u, 0x67452301u, 0xEFCDAB89u, false, false);
    x = fmaxf(x, __uint_as_float(p));
    u = __float_as_uint(x);  // xor 2
    p = __builtin_amdgcn_permlane16(u, u, 0x54761032u, 0xDCFE98BAu, false, false);
    x = fmaxf(x, __uint_as_float(p));
    u = __float_as_uint(x);  // xor 4
    p = __builtin_amdgcn_permlane16(u, u, 0x32107654u, 0xBA98FEDCu, false, false);
    x = fmaxf(x, __uint_as_float(p));
    u = __float_as_uint(x);  // xor 8
    p = __builtin_amdgcn_permlane16(u, u, 0xFEDCBA98u, 0x76543210u, false, false);
    x = fmaxf(x, __uint_as_float(p));
    return x;
#else
#pragma unroll
    for (int mask = 1; mask <= 8; mask <<= 1)
        x = fmaxf(x, __shfl_xor(x, mask, 32));
    return x;
#endif
}

// ---------------- prep: fp32 -> bf16 pack (K row-major, V transposed) --------
__global__ __launch_bounds__(256) void pack_kv(const float* __restrict__ k,
                                               const float* __restrict__ v,
                                               bf16_t* __restrict__ kb,
                                               bf16_t* __restrict__ vt)
{
    int idx = blockIdx.x * 256 + threadIdx.x;    // over N*D, d fastest
    int n = idx >> 6;
    int d = idx & 63;
    kb[idx] = (bf16_t)k[idx];
    vt[(size_t)d * N_CTX + n] = (bf16_t)v[idx];  // V^T: [d][n]
}

// ---------------- FlashAttention forward ------------------------------------
__global__ __launch_bounds__(NTHREADS) void fa_fwd_wmma_bf16(
    const float*  __restrict__ q,
    const bf16_t* __restrict__ Kg,   // [N][64] bf16
    const bf16_t* __restrict__ Vtg,  // [64][N] bf16 (V transposed)
    float* __restrict__ Oout,
    float* __restrict__ Lout)
{
    __shared__ bf16_t Klds[2][BC][KP];        // K tile, [kvrow][d], double-buffered
    __shared__ bf16_t Vt[2][DHEAD][KP];       // V tile transposed, [d][kvrow]
    __shared__ bf16_t Plds[WAVES][16][KP];    // per-wave P scratch [m][kvcol]

    const int t    = threadIdx.x;
    const int w    = t >> 5;
    const int lane = t & 31;
    const int g    = lane >> 4;
    const int nl   = lane & 15;
    const int qbase = blockIdx.x * BLOCK_Q + w * 16;
    // fold 1/sqrt(64) and log2(e) into Q so softmax runs in exp2 domain
    const float qscale = 0.125f * 1.44269504088896340736f;

    // ---- Q fragments (A-layout, pre-scaled) ----
    v16bf qf[2];
    {
        const float* qp = q + (size_t)(qbase + nl) * DHEAD;
#pragma unroll
        for (int c = 0; c < 2; ++c)
#pragma unroll
            for (int j = 0; j < 16; ++j) {
                int d = 32 * c + 16 * g + j;
                qf[c][j] = (bf16_t)(qp[d] * qscale);
            }
    }
    // all-ones B fragment for rowsum-by-WMMA
    v16bf onesf;
#pragma unroll
    for (int j = 0; j < 16; ++j) onesf[j] = (bf16_t)1.0f;

    v8f o[4];
#pragma unroll
    for (int nt = 0; nt < 4; ++nt)
#pragma unroll
        for (int r = 0; r < 8; ++r) o[nt][r] = 0.0f;
    float mrow[8], lrow[8];
#pragma unroll
    for (int r = 0; r < 8; ++r) { mrow[r] = -1.0e38f; lrow[r] = 0.0f; }

    // ---- prologue: stage tile 0 into buffer 0 ----
#pragma unroll
    for (int i = 0; i < 4; ++i) {
        int id = i * NTHREADS + t;           // 512 16B chunks per matrix
        int r = id >> 3, part = id & 7;
        cp_async16(Kg  + (size_t)r * DHEAD + part * 8, &Klds[0][r][part * 8]);
        cp_async16(Vtg + (size_t)r * N_CTX + part * 8, &Vt[0][r][part * 8]);
    }

    for (int kb = 0; kb < NT; ++kb) {
        const int buf = kb & 1;
        wait_async_all();          // this wave's copies for tile kb are in LDS
        __syncthreads();           // everyone's copies landed; prev compute done

        if (kb + 1 < NT) {         // overlap next-tile staging with compute
            const int nb = kb + 1;
#pragma unroll
            for (int i = 0; i < 4; ++i) {
                int id = i * NTHREADS + t;
                int r = id >> 3, part = id & 7;
                cp_async16(Kg  + (size_t)(nb * BC + r) * DHEAD + part * 8,
                           &Klds[buf ^ 1][r][part * 8]);
                cp_async16(Vtg + (size_t)r * N_CTX + nb * BC + part * 8,
                           &Vt[buf ^ 1][r][part * 8]);
            }
        }

        // ---- S' = (Q*qscale) @ K^T : 4 column tiles ----
        v8f s[4];
#pragma unroll
        for (int ct = 0; ct < 4; ++ct) {
            const bf16_t* krow = &Klds[buf][ct * 16 + nl][0];
            v16bf b0, b1;
#pragma unroll
            for (int j = 0; j < 16; ++j) {
                b0[j] = krow[16 * g + j];
                b1[j] = krow[32 + 16 * g + j];
            }
            v8f acc = {};
            acc = __builtin_amdgcn_wmma_f32_16x16x32_bf16(false, qf[0], false, b0,
                                                          (short)0, acc, false, false);
            acc = __builtin_amdgcn_wmma_f32_16x16x32_bf16(false, qf[1], false, b1,
                                                          (short)0, acc, false, false);
            s[ct] = acc;
        }

        // ---- row max (exact, needed for L) ----
        float rmax[8];
#pragma unroll
        for (int r = 0; r < 8; ++r)
            rmax[r] = fmaxf(fmaxf(s[0][r], s[1][r]), fmaxf(s[2][r], s[3][r]));
#pragma unroll
        for (int r = 0; r < 8; ++r)
            rmax[r] = rowmax_bfly16(rmax[r]);

        // ---- wave-uniform skip of alpha work when no row got a new max ----
        bool newmax = false;
#pragma unroll
        for (int r = 0; r < 8; ++r) newmax |= (rmax[r] > mrow[r]);
        if (__any((int)newmax)) {
            float alpha[8];
#pragma unroll
            for (int r = 0; r < 8; ++r) {
                float mnew = fmaxf(mrow[r], rmax[r]);
                alpha[r] = fast_exp2(mrow[r] - mnew);
                mrow[r] = mnew;
            }
#pragma unroll
            for (int r = 0; r < 8; ++r) lrow[r] *= alpha[r];
#pragma unroll
            for (int nt = 0; nt < 4; ++nt)
#pragma unroll
                for (int r = 0; r < 8; ++r)
                    o[nt][r] *= alpha[r];
        }

        // ---- P = 2^(S' - m'), write to wave-private LDS in bf16 ----
#pragma unroll
        for (int ct = 0; ct < 4; ++ct)
#pragma unroll
            for (int r = 0; r < 8; ++r) {
                float p = fast_exp2(s[ct][r] - mrow[r]);
                Plds[w][r + 8 * g][ct * 16 + nl] = (bf16_t)p;
            }

        // ---- P fragments (A-layout) from LDS ----
        v16bf pf[2];
        {
            const bf16_t* prow = &Plds[w][nl][0];
#pragma unroll
            for (int c = 0; c < 2; ++c)
#pragma unroll
                for (int j = 0; j < 16; ++j) {
                    int col = 32 * c + 16 * g + j;
                    pf[c][j] = prow[col];
                }
        }

        // ---- rowsum(P) via WMMA with ones: broadcast in C-layout ----
        {
            v8f rs = {};
            rs = __builtin_amdgcn_wmma_f32_16x16x32_bf16(false, pf[0], false, onesf,
                                                         (short)0, rs, false, false);
            rs = __builtin_amdgcn_wmma_f32_16x16x32_bf16(false, pf[1], false, onesf,
                                                         (short)0, rs, false, false);
#pragma unroll
            for (int r = 0; r < 8; ++r)
                lrow[r] += rs[r];
        }

        // ---- O += P @ V ----
#pragma unroll
        for (int nt = 0; nt < 4; ++nt) {
            const bf16_t* vrow = &Vt[buf][nt * 16 + nl][0];
            v16bf vb0, vb1;
#pragma unroll
            for (int j = 0; j < 16; ++j) {
                vb0[j] = vrow[16 * g + j];
                vb1[j] = vrow[32 + 16 * g + j];
            }
            o[nt] = __builtin_amdgcn_wmma_f32_16x16x32_bf16(false, pf[0], false, vb0,
                                                            (short)0, o[nt], false, false);
            o[nt] = __builtin_amdgcn_wmma_f32_16x16x32_bf16(false, pf[1], false, vb1,
                                                            (short)0, o[nt], false, false);
        }
    }

    // ---- epilogue ----
    float inv[8];
#pragma unroll
    for (int r = 0; r < 8; ++r) inv[r] = 1.0f / lrow[r];
#pragma unroll
    for (int nt = 0; nt < 4; ++nt)
#pragma unroll
        for (int r = 0; r < 8; ++r)
            Oout[(size_t)(qbase + r + 8 * g) * DHEAD + nt * 16 + nl] = o[nt][r] * inv[r];
    if (nl == 0) {
#pragma unroll
        for (int r = 0; r < 8; ++r)
            Lout[qbase + r + 8 * g] = lrow[r];
    }
}

extern "C" void kernel_launch(void* const* d_in, const int* in_sizes, int n_in,
                              void* d_out, int out_size, void* d_ws, size_t ws_size,
                              hipStream_t stream) {
    const float* q = (const float*)d_in[0];
    const float* k = (const float*)d_in[1];
    const float* v = (const float*)d_in[2];
    float* O = (float*)d_out;
    float* L = (float*)d_out + (size_t)N_CTX * DHEAD;

    bf16_t* Kg  = (bf16_t*)d_ws;                       // [N][64] bf16  (1 MB)
    bf16_t* Vtg = Kg + (size_t)N_CTX * DHEAD;          // [64][N] bf16  (1 MB)

    pack_kv<<<dim3((N_CTX * DHEAD) / 256), dim3(256), 0, stream>>>(k, v, Kg, Vtg);
    fa_fwd_wmma_bf16<<<dim3(N_CTX / BLOCK_Q), dim3(NTHREADS), 0, stream>>>(q, Kg, Vtg, O, L);
}
